// SchemaAwareModel_13846974562396
// MI455X (gfx1250) — matrix operand order repaired and checked by reading, not compiled, not run on hardware
//
#include <hip/hip_runtime.h>
#include <stdint.h>

// ---------------------------------------------------------------------------
// Problem constants (match reference: B,T,C,D,H = 64,32,32,1024,8)
// ---------------------------------------------------------------------------
#define B_  64
#define T_  32
#define C_  32
#define D_  1024
#define NH  8
#define DH  128          // D_/NH
#define SCALE 0.08838834764831845f   // 1/sqrt(128)

// ---------------------------------------------------------------------------
// WMMA fragment types (CDNA5 gfx1250, wave32, 16x16x32 bf16)
// ---------------------------------------------------------------------------
typedef __bf16 v16bf __attribute__((ext_vector_type(16)));
typedef float  v8f   __attribute__((ext_vector_type(8)));

struct alignas(16) U128 { unsigned int a, b, c, d; };   // trivial 16B chunk
union Frag { U128 q[2]; v16bf v; };                      // 32B = 16 x bf16

__device__ __forceinline__ unsigned short f2bf(float f) {
  unsigned int u = __float_as_uint(f);
  u += 0x7FFFu + ((u >> 16) & 1u);          // round-to-nearest-even
  return (unsigned short)(u >> 16);
}
__device__ __forceinline__ float bf2f(unsigned short h) {
  return __uint_as_float(((unsigned int)h) << 16);
}

// ---------------------------------------------------------------------------
// f32 -> bf16 row-major copy (A operands)
// ---------------------------------------------------------------------------
__global__ void k_cvt_bf16(const float* __restrict__ in,
                           unsigned short* __restrict__ out, int n) {
  int i = blockIdx.x * blockDim.x + threadIdx.x;
  if (i < n) out[i] = f2bf(in[i]);
}

// f32 [K,N] row-major -> bf16 [N,K] row-major (B operand = column-major K-contig)
__global__ void k_cvt_bf16_T(const float* __restrict__ in,
                             unsigned short* __restrict__ outT, int K, int N) {
  int i = blockIdx.x * blockDim.x + threadIdx.x;
  if (i < K * N) {
    int n = i / K, k = i % K;
    outT[i] = f2bf(in[(size_t)k * N + n]);
  }
}

// bias fold: out[n] = sum_k b_in[k] * W[k][n] + b_add[n]
__global__ void k_bias_fuse(const float* __restrict__ b_in,
                            const float* __restrict__ W,
                            const float* __restrict__ b_add,
                            float* __restrict__ out, int K, int N) {
  int n = blockIdx.x * blockDim.x + threadIdx.x;
  if (n >= N) return;
  float acc = b_add[n];
  for (int k = 0; k < K; ++k) acc += b_in[k] * W[(size_t)k * N + n];
  out[n] = acc;
}

// ---------------------------------------------------------------------------
// Generic WMMA bf16 GEMM:  C[M,N] = A[M,K] @ B[K,N] (+bias)
//   A   : bf16 row-major
//   BT  : bf16 [N,K] row-major (B column-major)
//   outF: optional f32 row-major,  outBT: optional bf16 [N,M] (transposed)
// One 16x16 tile per wave, 8 waves / block, grid = M/16 * N/16 / 8.
// ---------------------------------------------------------------------------
__global__ __launch_bounds__(256)
void k_gemm_bf16(const unsigned short* __restrict__ A,
                 const unsigned short* __restrict__ BT,
                 const float* __restrict__ bias,
                 float* __restrict__ outF,
                 unsigned short* __restrict__ outBT,
                 int M, int N, int K) {
  const int lane = threadIdx.x & 31;
  const int wave = threadIdx.x >> 5;
  const int half = lane >> 4;        // 0: lanes 0-15, 1: lanes 16-31
  const int lr   = lane & 15;
  const int mtiles = M >> 4;
  const int tile = blockIdx.x * 8 + wave;
  const int mt = tile % mtiles, nt = tile / mtiles;
  if (nt >= (N >> 4)) return;
  const int m0 = mt << 4, n0 = nt << 4;

  const unsigned short* arow = A  + (size_t)(m0 + lr) * K;
  const unsigned short* brow = BT + (size_t)(n0 + lr) * K;

  v8f acc = {0.f, 0.f, 0.f, 0.f, 0.f, 0.f, 0.f, 0.f};
  for (int k0 = 0; k0 < K; k0 += 32) {
    Frag a, b;
    // A 16-bit layout: lane-half 0 holds K[0..7]&[16..23], half 1 K[8..15]&[24..31]
    a.q[0] = *(const U128*)(arow + k0 + half * 8);
    a.q[1] = *(const U128*)(arow + k0 + 16 + half * 8);
    // B layout: lane-half 0 holds K[0..7]&[16..23], half 1 K[8..15]&[24..31]
    b.q[0] = *(const U128*)(brow + k0 + half * 8);
    b.q[1] = *(const U128*)(brow + k0 + 16 + half * 8);
    acc = __builtin_amdgcn_wmma_f32_16x16x32_bf16(false, a.v, false, b.v,
                                                  (short)0, acc, false, false);
  }
  const int colC  = n0 + lr;
  const int rbase = m0 + (half << 3);
  const float bv = bias ? bias[colC] : 0.f;
  for (int j = 0; j < 8; ++j) {
    float v = acc[j] + bv;
    if (outF)  outF[(size_t)(rbase + j) * N + colC] = v;
    if (outBT) outBT[(size_t)colC * M + rbase + j] = f2bf(v);  // contiguous
  }
}

// ---------------------------------------------------------------------------
// Stage 2: fused per-(b,t) column attention + residual + LayerNorm.
// One block (256 thr = 8 waves) per (b,t) group; dynamic LDS = 210176 B.
//   kh = col_g @ WckT + bck,  vh = col_g @ WcvT + bcv   (WMMA, LDS A, bf16 out)
//   qh = table_p_row @ Wq1 + bq1 (VALU GEMV)
//   8 heads x 32 keys masked softmax, AV, @Wo1, +residual, LN -> enhA (bf16)
// ---------------------------------------------------------------------------
__global__ __launch_bounds__(256)
void k_group_attn(const float* __restrict__ col,
                  const unsigned char* __restrict__ col_mask,
                  const float* __restrict__ table_p,
                  const unsigned short* __restrict__ WckT,
                  const float* __restrict__ bck,
                  const unsigned short* __restrict__ WcvT,
                  const float* __restrict__ bcv,
                  const float* __restrict__ Wq1, const float* __restrict__ bq1,
                  const float* __restrict__ Wo1, const float* __restrict__ bo1,
                  const float* __restrict__ g1, const float* __restrict__ be1,
                  unsigned short* __restrict__ enhA) {
  extern __shared__ char smem[];
  unsigned short* colA = (unsigned short*)(smem);            // 32x1024 bf16 64KB
  unsigned short* khS  = (unsigned short*)(smem + 65536);    // 64KB
  unsigned short* vhS  = (unsigned short*)(smem + 131072);   // 64KB
  float* qv   = (float*)(smem + 196608);                     // table_p row 4KB
  float* qh   = (float*)(smem + 200704);                     // 4KB
  float* sc   = (float*)(smem + 204800);                     // 8x32 scores 1KB
  float* obuf = (float*)(smem + 205824);                     // 4KB
  float* red  = (float*)(smem + 209920);                     // reductions

  const int bt   = blockIdx.x;            // 0 .. B*T-1
  const int tid  = threadIdx.x;
  const int lane = tid & 31;
  const int wave = tid >> 5;
  const int half = lane >> 4;
  const int lr   = lane & 15;

  // ---- load col[b,t] -> bf16 LDS; load table_p row ----
  const float4* cg = (const float4*)(col + (size_t)bt * C_ * D_);
  for (int i = tid; i < C_ * D_ / 4; i += 256) {
    float4 v = cg[i];
    int e = i * 4;
    colA[e]     = f2bf(v.x); colA[e + 1] = f2bf(v.y);
    colA[e + 2] = f2bf(v.z); colA[e + 3] = f2bf(v.w);
  }
  for (int i = tid; i < D_; i += 256) qv[i] = table_p[(size_t)bt * D_ + i];
  __syncthreads();

  // ---- qh = qv @ Wq1 + bq1 (coalesced float4 weight reads) ----
  {
    const int n0 = tid * 4;
    float a0 = bq1[n0], a1 = bq1[n0 + 1], a2 = bq1[n0 + 2], a3 = bq1[n0 + 3];
    const float4* W = (const float4*)Wq1;
    for (int k = 0; k < D_; ++k) {
      float qk = qv[k];
      float4 w = W[k * (D_ / 4) + tid];
      a0 += qk * w.x; a1 += qk * w.y; a2 += qk * w.z; a3 += qk * w.w;
    }
    qh[n0] = a0; qh[n0 + 1] = a1; qh[n0 + 2] = a2; qh[n0 + 3] = a3;
  }

  // ---- kh / vh via WMMA: [32,1024] = colA @ W{ck,cv}; 128 tiles each ----
  for (int g = 0; g < 2; ++g) {
    const unsigned short* BT = g ? WcvT : WckT;
    const float* bias = g ? bcv : bck;
    unsigned short* dst = g ? vhS : khS;
    for (int t = wave; t < 128; t += 8) {
      const int m0 = (t & 1) << 4;
      const int n0 = (t >> 1) << 4;
      const unsigned short* arow = colA + (size_t)(m0 + lr) * D_;
      const unsigned short* brow = BT + (size_t)(n0 + lr) * D_;
      v8f acc = {0.f, 0.f, 0.f, 0.f, 0.f, 0.f, 0.f, 0.f};
      for (int k0 = 0; k0 < D_; k0 += 32) {
        Frag a, b;
        a.q[0] = *(const U128*)(arow + k0 + half * 8);
        a.q[1] = *(const U128*)(arow + k0 + 16 + half * 8);
        b.q[0] = *(const U128*)(brow + k0 + half * 8);
        b.q[1] = *(const U128*)(brow + k0 + 16 + half * 8);
        acc = __builtin_amdgcn_wmma_f32_16x16x32_bf16(false, a.v, false, b.v,
                                                      (short)0, acc, false, false);
      }
      const int colC = n0 + lr;
      const float bv = bias[colC];
      const int rbase = m0 + (half << 3);
      for (int j = 0; j < 8; ++j)
        dst[(size_t)(rbase + j) * D_ + colC] = f2bf(acc[j] + bv);
    }
  }
  __syncthreads();

  // ---- scores: one (head, key) per thread ----
  {
    const int h = tid >> 5, j = tid & 31;
    const unsigned short* krow = khS + (size_t)j * D_ + h * DH;
    const float* qhh = qh + h * DH;
    float s = 0.f;
    for (int d = 0; d < DH; ++d) s += qhh[d] * bf2f(krow[d]);
    s *= SCALE;
    sc[tid] = col_mask[(size_t)bt * C_ + j] ? s : -1e30f;
  }
  __syncthreads();
  if (tid < NH) {                               // softmax per head (32 keys)
    float* row = sc + tid * 32;
    float mx = -1e30f;
    for (int j = 0; j < 32; ++j) mx = fmaxf(mx, row[j]);
    float sum = 0.f;
    for (int j = 0; j < 32; ++j) { float e = __expf(row[j] - mx); row[j] = e; sum += e; }
    float inv = 1.f / sum;
    for (int j = 0; j < 32; ++j) row[j] *= inv;
  }
  __syncthreads();

  // ---- o = a @ vh ----
  {
    const int d0 = tid * 4, h = d0 >> 7;
    float o0 = 0.f, o1 = 0.f, o2 = 0.f, o3 = 0.f;
    for (int j = 0; j < C_; ++j) {
      float aj = sc[h * 32 + j];
      const unsigned short* vrow = vhS + (size_t)j * D_ + d0;
      o0 += aj * bf2f(vrow[0]); o1 += aj * bf2f(vrow[1]);
      o2 += aj * bf2f(vrow[2]); o3 += aj * bf2f(vrow[3]);
    }
    obuf[d0] = o0; obuf[d0 + 1] = o1; obuf[d0 + 2] = o2; obuf[d0 + 3] = o3;
  }
  __syncthreads();

  // ---- attn = o @ Wo1 + bo1; pre = table_p + attn; LayerNorm -> enhA ----
  {
    const int n0 = tid * 4;
    float a0 = bo1[n0], a1 = bo1[n0 + 1], a2 = bo1[n0 + 2], a3 = bo1[n0 + 3];
    const float4* W = (const float4*)Wo1;
    for (int d = 0; d < D_; ++d) {
      float od = obuf[d];
      float4 w = W[d * (D_ / 4) + tid];
      a0 += od * w.x; a1 += od * w.y; a2 += od * w.z; a3 += od * w.w;
    }
    float p0 = a0 + qv[n0],     p1 = a1 + qv[n0 + 1];
    float p2 = a2 + qv[n0 + 2], p3 = a3 + qv[n0 + 3];
    float s1 = p0 + p1 + p2 + p3;
    float s2 = p0 * p0 + p1 * p1 + p2 * p2 + p3 * p3;
    for (int off = 16; off > 0; off >>= 1) {
      s1 += __shfl_down(s1, off);
      s2 += __shfl_down(s2, off);
    }
    if (lane == 0) { red[wave] = s1; red[8 + wave] = s2; }
    __syncthreads();
    float t1 = 0.f, t2 = 0.f;
    for (int i = 0; i < 8; ++i) { t1 += red[i]; t2 += red[8 + i]; }
    float mean = t1 * (1.f / D_);
    float var  = t2 * (1.f / D_) - mean * mean;
    float inv  = rsqrtf(var + 1e-5f);
    unsigned short* dst = enhA + (size_t)bt * D_ + n0;
    dst[0] = f2bf((p0 - mean) * inv * g1[n0]     + be1[n0]);
    dst[1] = f2bf((p1 - mean) * inv * g1[n0 + 1] + be1[n0 + 1]);
    dst[2] = f2bf((p2 - mean) * inv * g1[n0 + 2] + be1[n0 + 2]);
    dst[3] = f2bf((p3 - mean) * inv * g1[n0 + 3] + be1[n0 + 3]);
  }
}

// ---------------------------------------------------------------------------
// Stage 3: per-batch query attention over T tables + LN2 + final matmul.
// One block per batch element. GEMV-shaped -> f32 VALU with coalesced weights.
// ---------------------------------------------------------------------------
__global__ __launch_bounds__(256)
void k_attn2(const float* __restrict__ q,
             const unsigned char* __restrict__ table_mask,
             const float* __restrict__ Wql, const float* __restrict__ bql,
             const float* __restrict__ Wq2, const float* __restrict__ bq2,
             const float* __restrict__ kh2, const float* __restrict__ vh2,
             const float* __restrict__ Wo2, const float* __restrict__ bo2,
             const float* __restrict__ g2,  const float* __restrict__ be2,
             const float* __restrict__ Wout, const float* __restrict__ bout,
             float* __restrict__ out) {
  __shared__ float qrow[D_], q1[D_], qh2s[D_], sc2[256], obuf[D_], q2s[D_], red[16];
  const int b = blockIdx.x, tid = threadIdx.x;
  const int lane = tid & 31, wave = tid >> 5;
  const int n0 = tid * 4;

  for (int i = tid; i < D_; i += 256) qrow[i] = q[(size_t)b * D_ + i];
  __syncthreads();

  // q1 = qrow @ Wql + bql
  {
    float a0 = bql[n0], a1 = bql[n0 + 1], a2 = bql[n0 + 2], a3 = bql[n0 + 3];
    const float4* W = (const float4*)Wql;
    for (int k = 0; k < D_; ++k) {
      float qk = qrow[k]; float4 w = W[k * (D_ / 4) + tid];
      a0 += qk * w.x; a1 += qk * w.y; a2 += qk * w.z; a3 += qk * w.w;
    }
    q1[n0] = a0; q1[n0 + 1] = a1; q1[n0 + 2] = a2; q1[n0 + 3] = a3;
  }
  __syncthreads();

  // qh2 = q1 @ Wq2 + bq2
  {
    float a0 = bq2[n0], a1 = bq2[n0 + 1], a2 = bq2[n0 + 2], a3 = bq2[n0 + 3];
    const float4* W = (const float4*)Wq2;
    for (int k = 0; k < D_; ++k) {
      float qk = q1[k]; float4 w = W[k * (D_ / 4) + tid];
      a0 += qk * w.x; a1 += qk * w.y; a2 += qk * w.z; a3 += qk * w.w;
    }
    qh2s[n0] = a0; qh2s[n0 + 1] = a1; qh2s[n0 + 2] = a2; qh2s[n0 + 3] = a3;
  }
  __syncthreads();

  // masked scores over T_ keys
  {
    const int h = tid >> 5, t = tid & 31;
    const float* krow = kh2 + ((size_t)b * T_ + t) * D_ + h * DH;
    const float* qhh = qh2s + h * DH;
    float s = 0.f;
    for (int d = 0; d < DH; ++d) s += qhh[d] * krow[d];
    s *= SCALE;
    sc2[tid] = table_mask[(size_t)b * T_ + t] ? s : -1e30f;
  }
  __syncthreads();
  if (tid < NH) {
    float* row = sc2 + tid * 32;
    float mx = -1e30f;
    for (int j = 0; j < 32; ++j) mx = fmaxf(mx, row[j]);
    float sum = 0.f;
    for (int j = 0; j < 32; ++j) { float e = __expf(row[j] - mx); row[j] = e; sum += e; }
    float inv = 1.f / sum;
    for (int j = 0; j < 32; ++j) row[j] *= inv;
  }
  __syncthreads();

  // o = a @ vh2
  {
    const int d0 = tid * 4, h = d0 >> 7;
    float o0 = 0.f, o1 = 0.f, o2 = 0.f, o3 = 0.f;
    for (int t = 0; t < T_; ++t) {
      float at = sc2[h * 32 + t];
      const float* vrow = vh2 + ((size_t)b * T_ + t) * D_ + d0;
      o0 += at * vrow[0]; o1 += at * vrow[1]; o2 += at * vrow[2]; o3 += at * vrow[3];
    }
    obuf[d0] = o0; obuf[d0 + 1] = o1; obuf[d0 + 2] = o2; obuf[d0 + 3] = o3;
  }
  __syncthreads();

  // attn2 = o @ Wo2 + bo2; pre = q1 + attn2; LN2 -> q2s
  {
    float a0 = bo2[n0], a1 = bo2[n0 + 1], a2 = bo2[n0 + 2], a3 = bo2[n0 + 3];
    const float4* W = (const float4*)Wo2;
    for (int d = 0; d < D_; ++d) {
      float od = obuf[d]; float4 w = W[d * (D_ / 4) + tid];
      a0 += od * w.x; a1 += od * w.y; a2 += od * w.z; a3 += od * w.w;
    }
    float p0 = a0 + q1[n0],     p1 = a1 + q1[n0 + 1];
    float p2 = a2 + q1[n0 + 2], p3 = a3 + q1[n0 + 3];
    float s1 = p0 + p1 + p2 + p3;
    float s2 = p0 * p0 + p1 * p1 + p2 * p2 + p3 * p3;
    for (int off = 16; off > 0; off >>= 1) {
      s1 += __shfl_down(s1, off);
      s2 += __shfl_down(s2, off);
    }
    if (lane == 0) { red[wave] = s1; red[8 + wave] = s2; }
    __syncthreads();
    float t1 = 0.f, t2 = 0.f;
    for (int i = 0; i < 8; ++i) { t1 += red[i]; t2 += red[8 + i]; }
    float mean = t1 * (1.f / D_);
    float var  = t2 * (1.f / D_) - mean * mean;
    float inv  = rsqrtf(var + 1e-5f);
    q2s[n0]     = (p0 - mean) * inv * g2[n0]     + be2[n0];
    q2s[n0 + 1] = (p1 - mean) * inv * g2[n0 + 1] + be2[n0 + 1];
    q2s[n0 + 2] = (p2 - mean) * inv * g2[n0 + 2] + be2[n0 + 2];
    q2s[n0 + 3] = (p3 - mean) * inv * g2[n0 + 3] + be2[n0 + 3];
  }
  __syncthreads();

  // out = q2 @ Wout + bout
  {
    float a0 = bout[n0], a1 = bout[n0 + 1], a2 = bout[n0 + 2], a3 = bout[n0 + 3];
    const float4* W = (const float4*)Wout;
    for (int k = 0; k < D_; ++k) {
      float qk = q2s[k]; float4 w = W[k * (D_ / 4) + tid];
      a0 += qk * w.x; a1 += qk * w.y; a2 += qk * w.z; a3 += qk * w.w;
    }
    float* o = out + (size_t)b * D_ + n0;
    o[0] = a0; o[1] = a1; o[2] = a2; o[3] = a3;
  }
}

// ---------------------------------------------------------------------------
// Host-side orchestration
// ---------------------------------------------------------------------------
extern "C" void kernel_launch(void* const* d_in, const int* in_sizes, int n_in,
                              void* d_out, int out_size, void* d_ws, size_t ws_size,
                              hipStream_t stream) {
  (void)in_sizes; (void)n_in; (void)out_size; (void)ws_size;
  // inputs in setup_inputs() dict order
  const float* q          = (const float*)d_in[0];
  const float* table      = (const float*)d_in[1];
  const float* col        = (const float*)d_in[2];
  const unsigned char* table_mask = (const unsigned char*)d_in[3];
  const unsigned char* col_mask   = (const unsigned char*)d_in[4];
  const float* Wt  = (const float*)d_in[5];
  const float* Wc  = (const float*)d_in[6];
  const float* Wql = (const float*)d_in[7];
  const float* Wout= (const float*)d_in[8];
  const float* Wq1 = (const float*)d_in[9];
  const float* Wk1 = (const float*)d_in[10];
  const float* Wv1 = (const float*)d_in[11];
  const float* Wo1 = (const float*)d_in[12];
  const float* Wq2 = (const float*)d_in[13];
  const float* Wk2 = (const float*)d_in[14];
  const float* Wv2 = (const float*)d_in[15];
  const float* Wo2 = (const float*)d_in[16];
  const float* bt  = (const float*)d_in[17];
  const float* bc  = (const float*)d_in[18];
  const float* bql = (const float*)d_in[19];
  const float* bout= (const float*)d_in[20];
  const float* bq1 = (const float*)d_in[21];
  const float* bk1 = (const float*)d_in[22];
  const float* bv1 = (const float*)d_in[23];
  const float* bo1 = (const float*)d_in[24];
  const float* bq2 = (const float*)d_in[25];
  const float* bk2 = (const float*)d_in[26];
  const float* bv2 = (const float*)d_in[27];
  const float* bo2 = (const float*)d_in[28];
  const float* g1  = (const float*)d_in[29];
  const float* be1 = (const float*)d_in[30];
  const float* g2  = (const float*)d_in[31];
  const float* be2 = (const float*)d_in[32];
  (void)bc;

  // workspace carve-up (256B aligned slabs)
  char* ws = (char*)d_ws;
  size_t off = 0;
  auto alloc = [&](size_t bytes) -> char* {
    char* p = ws + off;
    off += (bytes + 255) & ~(size_t)255;
    return p;
  };
  const size_t DD2 = (size_t)D_ * D_ * 2;                 // bf16 DxD
  unsigned short* tableA = (unsigned short*)alloc((size_t)B_ * T_ * D_ * 2);
  float*          tableP = (float*)         alloc((size_t)B_ * T_ * D_ * 4);
  unsigned short* WcA    = (unsigned short*)alloc(DD2);
  unsigned short* WtT    = (unsigned short*)alloc(DD2);
  unsigned short* Wk1T   = (unsigned short*)alloc(DD2);
  unsigned short* Wv1T   = (unsigned short*)alloc(DD2);
  unsigned short* Wk2T   = (unsigned short*)alloc(DD2);
  unsigned short* Wv2T   = (unsigned short*)alloc(DD2);
  unsigned short* WckT   = (unsigned short*)alloc(DD2);
  unsigned short* WcvT   = (unsigned short*)alloc(DD2);
  float*          bck    = (float*)alloc(D_ * 4);
  float*          bcv    = (float*)alloc(D_ * 4);
  unsigned short* enhA   = (unsigned short*)alloc((size_t)B_ * T_ * D_ * 2);
  float*          kh2    = (float*)alloc((size_t)B_ * T_ * D_ * 4);
  float*          vh2    = (float*)alloc((size_t)B_ * T_ * D_ * 4);

  const int thr = 256;
  const int nTab = B_ * T_ * D_;       // 2M
  const int nW   = D_ * D_;            // 1M

  // 1) precision/layout conversions
  k_cvt_bf16  <<<(nTab + thr - 1) / thr, thr, 0, stream>>>(table, tableA, nTab);
  k_cvt_bf16  <<<(nW + thr - 1) / thr,   thr, 0, stream>>>(Wc, WcA, nW);
  k_cvt_bf16_T<<<(nW + thr - 1) / thr,   thr, 0, stream>>>(Wt,  WtT,  D_, D_);
  k_cvt_bf16_T<<<(nW + thr - 1) / thr,   thr, 0, stream>>>(Wk1, Wk1T, D_, D_);
  k_cvt_bf16_T<<<(nW + thr - 1) / thr,   thr, 0, stream>>>(Wv1, Wv1T, D_, D_);
  k_cvt_bf16_T<<<(nW + thr - 1) / thr,   thr, 0, stream>>>(Wk2, Wk2T, D_, D_);
  k_cvt_bf16_T<<<(nW + thr - 1) / thr,   thr, 0, stream>>>(Wv2, Wv2T, D_, D_);

  // 2) fold weight products (col_p never materialized): Wck = Wc@Wk1, Wcv = Wc@Wv1
  k_bias_fuse<<<D_ / thr, thr, 0, stream>>>(bc, Wk1, bk1, bck, D_, D_);
  k_bias_fuse<<<D_ / thr, thr, 0, stream>>>(bc, Wv1, bv1, bcv, D_, D_);
  k_gemm_bf16<<<(D_ / 16) * (D_ / 16) / 8, thr, 0, stream>>>(
      WcA, Wk1T, nullptr, nullptr, WckT, D_, D_, D_);
  k_gemm_bf16<<<(D_ / 16) * (D_ / 16) / 8, thr, 0, stream>>>(
      WcA, Wv1T, nullptr, nullptr, WcvT, D_, D_, D_);

  // 3) table_p = table @ Wt + bt
  k_gemm_bf16<<<(B_ * T_ / 16) * (D_ / 16) / 8, thr, 0, stream>>>(
      tableA, WtT, bt, tableP, nullptr, B_ * T_, D_, D_);

  // 4) fused per-(b,t) column attention + LN1 -> enhanced (bf16 A layout)
  k_group_attn<<<B_ * T_, thr, 210176, stream>>>(
      col, col_mask, tableP, WckT, bck, WcvT, bcv,
      Wq1, bq1, Wo1, bo1, g1, be1, enhA);

  // 5) kh2/vh2 projections of enhanced
  k_gemm_bf16<<<(B_ * T_ / 16) * (D_ / 16) / 8, thr, 0, stream>>>(
      enhA, Wk2T, bk2, kh2, nullptr, B_ * T_, D_, D_);
  k_gemm_bf16<<<(B_ * T_ / 16) * (D_ / 16) / 8, thr, 0, stream>>>(
      enhA, Wv2T, bv2, vh2, nullptr, B_ * T_, D_, D_);

  // 6) second attention + LN2 + final matmul -> out [B, D]
  k_attn2<<<B_, thr, 0, stream>>>(q, table_mask, Wql, bql, Wq2, bq2,
                                  kh2, vh2, Wo2, bo2, g2, be2,
                                  Wout, bout, (float*)d_out);
}
